// VSSBlock_42812234006786
// MI455X (gfx1250) — compile-verified, matrix operand order checked by reading, not run on hardware
//
#include <hip/hip_runtime.h>
#include <hip/hip_bf16.h>
#include <math.h>

// ---------------------------------------------------------------------------
// VSS block (VMamba) for MI455X / gfx1250.
// B=2, C=128, H=W=64, D_INNER=256, D_STATE=16, DT_RANK=8, K=4 directions.
// GEMMs (in_proj / x_proj / out_proj) use v_wmma_f32_16x16x32_f16 with
// multi-N-tile A-fragment reuse; all fragment loads are contiguous f16
// (vectorize to global_load_b128). Selective scan: 16 lanes per (b,k,d)
// sequence (one lane per state n), shfl-xor reduction, atomic scatter into
// the direction-combined output. Scan directions never materialized.
// ---------------------------------------------------------------------------

typedef __attribute__((ext_vector_type(16))) _Float16 v16h;
typedef __attribute__((ext_vector_type(8)))  float    v8f;

#define BB   2
#define CC   128
#define LL   4096           // H*W
#define MM   8192           // B*L
#define DI   256            // D_INNER
#define DSN  16             // D_STATE
#define RK   8              // DT_RANK
#define KD   4              // scan directions
#define CMID 42             // C // 3
#define CSQ  4              // C // 30

__device__ __forceinline__ int frag_k(int lane, int i) {
  // 16-bit A/B fragment layout (ISA 7.12.2): lanes 0-15 hold K 0-7 & 16-23,
  // lanes 16-31 hold K 8-15 & 24-31.
  return ((i < 8) ? i : (i + 8)) + ((lane & 16) ? 8 : 0);
}
__device__ __forceinline__ float siluf(float x) { return x / (1.f + __expf(-x)); }
__device__ __forceinline__ float softplusf(float x) { return (x > 20.f) ? x : log1pf(__expf(x)); }
__device__ __forceinline__ float geluf(float x) { return 0.5f * x * (1.f + erff(x * 0.70710678118f)); }

// scan index l of direction k  ->  linear h*64+w spatial position
__device__ __forceinline__ int scan_pos(int k, int l) {
  int ll = (k >= 2) ? (LL - 1 - l) : l;
  if (k & 1) { int h = ll & 63; int w = ll >> 6; return h * 64 + w; }  // w-major scan
  return ll;
}

// ---------------------------------------------------------------- helpers ---
__global__ void k_f32_to_f16(const float* __restrict__ s, _Float16* __restrict__ d, int n) {
  int i = blockIdx.x * blockDim.x + threadIdx.x;
  if (i < n) d[i] = (_Float16)s[i];
}
__global__ void k_zero(float* __restrict__ p, int n) {
  int i = blockIdx.x * blockDim.x + threadIdx.x;
  if (i < n) p[i] = 0.f;
}

// ------------------------------------------------ LN1 over C (per pixel) ---
__global__ void k_ln1(const float* __restrict__ input, const float* __restrict__ g,
                      const float* __restrict__ bt, _Float16* __restrict__ xn) {
  int m = blockIdx.x * blockDim.x + threadIdx.x;
  if (m >= MM) return;
  int b = m >> 12, hw = m & 4095;
  const float* base = input + (size_t)b * CC * LL + hw;     // (B,C,H,W), stride L per channel
  float mu = 0.f;
  for (int c = 0; c < CC; ++c) mu += base[c * LL];
  mu *= (1.f / CC);
  float var = 0.f;
  for (int c = 0; c < CC; ++c) { float t = base[c * LL] - mu; var += t * t; }
  var *= (1.f / CC);
  float inv = rsqrtf(var + 1e-5f);
  for (int c = 0; c < CC; ++c)
    xn[(size_t)m * CC + c] = (_Float16)((base[c * LL] - mu) * inv * g[c] + bt[c]);
}

// -------------------------------- GEMM1: xz = Xn(8192x128) @ Win^T(128x512)
// one wave = 16x64 strip (4 N-tiles share each A fragment).
// split on the fly: cols 0..255 -> xc (B,D,H,W), cols 256..511 -> z (M,256)
__global__ void k_inproj(const _Float16* __restrict__ xn, const _Float16* __restrict__ wf,
                         float* __restrict__ xc, float* __restrict__ z) {
  int wave = (blockIdx.x * blockDim.x + threadIdx.x) >> 5;   // 0..4095
  int lane = threadIdx.x & 31;
  int tm = wave >> 3;                  // 512 M-tiles
  int tng = wave & 7;                  // 8 groups of 4 N-tiles (N=512)
  int mrow = tm * 16 + (lane & 15);
  v8f acc[4] = {{}, {}, {}, {}};
  for (int kb = 0; kb < CC; kb += 32) {
    v16h a;
#pragma unroll
    for (int i = 0; i < 16; ++i) a[i] = xn[(size_t)mrow * CC + kb + frag_k(lane, i)];
#pragma unroll
    for (int nt = 0; nt < 4; ++nt) {
      int ncol = (tng * 4 + nt) * 16 + (lane & 15);
      v16h bfr;
#pragma unroll
      for (int i = 0; i < 16; ++i) bfr[i] = wf[(size_t)ncol * CC + kb + frag_k(lane, i)];
      acc[nt] = __builtin_amdgcn_wmma_f32_16x16x32_f16(false, a, false, bfr, (short)0, acc[nt], false, false);
    }
  }
#pragma unroll
  for (int nt = 0; nt < 4; ++nt)
#pragma unroll
    for (int r = 0; r < 8; ++r) {
      int row = tm * 16 + r + ((lane & 16) ? 8 : 0);
      int col = (tng * 4 + nt) * 16 + (lane & 15);
      int b = row >> 12, hw = row & 4095;
      float v = acc[nt][r];
      if (col < DI) xc[((size_t)b * DI + col) * LL + hw] = v;
      else          z[(size_t)row * DI + (col - DI)] = v;
    }
}

// -------- depthwise 3x3 conv + bias + SiLU; also emit channel-contiguous
// f16 transpose XCAT[(b*L+hw)*DI+d] so x_proj A-fragments load contiguously.
__global__ void k_dwconv(const float* __restrict__ xc, const float* __restrict__ w,
                         const float* __restrict__ bias, float* __restrict__ xca,
                         _Float16* __restrict__ xcat) {
  int idx = blockIdx.x * blockDim.x + threadIdx.x;       // (b,d,hw)
  if (idx >= BB * DI * LL) return;
  int hw = idx & 4095, d = (idx >> 12) & 255, b = idx >> 20;
  int h = hw >> 6, wq = hw & 63;
  float s = bias[d];
  const float* src = xc + ((size_t)b * DI + d) * LL;
#pragma unroll
  for (int i = 0; i < 3; ++i)
#pragma unroll
    for (int j = 0; j < 3; ++j) {
      int hh = h + i - 1, ww = wq + j - 1;
      if ((unsigned)hh < 64u && (unsigned)ww < 64u)
        s += src[hh * 64 + ww] * w[d * 9 + i * 3 + j];
    }
  float act = siluf(s);
  xca[idx] = act;
  xcat[((size_t)b * LL + hw) * DI + d] = (_Float16)act;
}

// ----------------- GEMM2: x_dbl[b,k,c,l] = sum_d xs[b,k,d,l]*x_proj_w[k,c,d]
// per (b,k): A = xs^T (Lx256) from XCAT via scan_pos (contiguous f16 in d),
// B = Wxp (40x256 padded to 48). One wave = full 16x48 strip (3 N-tiles).
__global__ void k_xproj(const _Float16* __restrict__ xcat, const _Float16* __restrict__ wxp,
                        float* __restrict__ xdbl) {
  int bk = blockIdx.y;                 // 0..7
  int b = bk >> 2, k = bk & 3;
  int tm = (blockIdx.x * blockDim.x + threadIdx.x) >> 5;     // 0..255 L-tiles
  int lane = threadIdx.x & 31;
  int mrow = tm * 16 + (lane & 15);    // scan index l
  int hwp = scan_pos(k, mrow);
  const _Float16* abase = xcat + ((size_t)b * LL + hwp) * DI;
  v8f acc[3] = {{}, {}, {}};
  for (int kb = 0; kb < DI; kb += 32) {
    v16h a;
#pragma unroll
    for (int i = 0; i < 16; ++i) a[i] = abase[kb + frag_k(lane, i)];
#pragma unroll
    for (int nt = 0; nt < 3; ++nt) {
      int ncol = nt * 16 + (lane & 15);            // c in 0..47
      v16h bfr = {};
      if (ncol < 40) {
#pragma unroll
        for (int i = 0; i < 16; ++i) bfr[i] = wxp[((size_t)k * 40 + ncol) * DI + kb + frag_k(lane, i)];
      }
      acc[nt] = __builtin_amdgcn_wmma_f32_16x16x32_f16(false, a, false, bfr, (short)0, acc[nt], false, false);
    }
  }
#pragma unroll
  for (int nt = 0; nt < 3; ++nt)
#pragma unroll
    for (int r = 0; r < 8; ++r) {
      int l = tm * 16 + r + ((lane & 16) ? 8 : 0);
      int c = nt * 16 + (lane & 15);
      if (c < 40) xdbl[(((size_t)b * KD + k) * 40 + c) * LL + l] = acc[nt][r];
    }
}

// --------------------------------------------------- selective scan + Ds ---
// 16 lanes per (b,k,d) sequence; lane = state index n. dt recomputed on the
// fly from dt_rank=8 projection; y reduced over lanes, scattered with
// global_atomic_add at the direction-mapped spatial position.
__global__ void k_scan(const float* __restrict__ xca, const float* __restrict__ xdbl,
                       const float* __restrict__ dt_w, const float* __restrict__ dt_b,
                       const float* __restrict__ A_logs, const float* __restrict__ Ds,
                       float* __restrict__ ycomb) {
  int gid = blockIdx.x * blockDim.x + threadIdx.x;
  int n = gid & 15;
  int seq = gid >> 4;
  if (seq >= BB * KD * DI) return;
  int d = seq & 255, k = (seq >> 8) & 3, b = seq >> 10;
  float A = -__expf(A_logs[((size_t)k * DI + d) * DSN + n]);
  float Dval = Ds[k * DI + d];
  float dtw[RK];
#pragma unroll
  for (int r = 0; r < RK; ++r) dtw[r] = dt_w[((size_t)k * DI + d) * RK + r];
  float dtb = dt_b[k * DI + d];
  const float* xd = xdbl + ((size_t)b * KD + k) * 40 * LL;
  const float* xrow = xca + ((size_t)b * DI + d) * LL;
  float h = 0.f;
  for (int l = 0; l < LL; ++l) {
    __builtin_prefetch(&xd[l + 128], 0, 1);               // global_prefetch_b8
    float dtr = dtb;
#pragma unroll
    for (int r = 0; r < RK; ++r) dtr += dtw[r] * xd[(size_t)r * LL + l];
    float dt = softplusf(dtr);
    int pos = scan_pos(k, l);
    float x  = xrow[pos];
    float Bn = xd[(size_t)(8 + n) * LL + l];
    float Cn = xd[(size_t)(24 + n) * LL + l];
    h = __expf(dt * A) * h + dt * x * Bn;
    float y = h * Cn;
    y += __shfl_xor(y, 8, 32);
    y += __shfl_xor(y, 4, 32);
    y += __shfl_xor(y, 2, 32);
    y += __shfl_xor(y, 1, 32);
    if (n == 0)
      atomicAdd(&ycomb[((size_t)b * LL + pos) * DI + d], y + Dval * x);
  }
}

// ------------------------- out_norm (LN over 256) + silu(z) gate -> f16 ---
__global__ void k_gate(const float* __restrict__ yc, const float* __restrict__ z,
                       const float* __restrict__ g, const float* __restrict__ bt,
                       _Float16* __restrict__ yg) {
  int m = blockIdx.x * blockDim.x + threadIdx.x;
  if (m >= MM) return;
  const float* row = yc + (size_t)m * DI;
  float mu = 0.f;
  for (int c = 0; c < DI; ++c) mu += row[c];
  mu *= (1.f / DI);
  float var = 0.f;
  for (int c = 0; c < DI; ++c) { float t = row[c] - mu; var += t * t; }
  var *= (1.f / DI);
  float inv = rsqrtf(var + 1e-5f);
  const float* zr = z + (size_t)m * DI;
  for (int c = 0; c < DI; ++c) {
    float v = (row[c] - mu) * inv * g[c] + bt[c];
    yg[(size_t)m * DI + c] = (_Float16)(v * siluf(zr[c]));
  }
}

// -------- GEMM3: out_proj (8192x256 @ 256x128) + fused skip residual ------
// one wave = 16x64 strip (4 N-tiles share each A fragment).
__global__ void k_outproj(const _Float16* __restrict__ yg, const _Float16* __restrict__ wf,
                          const float* __restrict__ input, const float* __restrict__ skip,
                          float* __restrict__ xafter) {
  int wave = (blockIdx.x * blockDim.x + threadIdx.x) >> 5;   // 0..1023
  int lane = threadIdx.x & 31;
  int tm = wave >> 1;                  // 512 M-tiles
  int tng = wave & 1;                  // 2 groups of 4 N-tiles (N=128)
  int mrow = tm * 16 + (lane & 15);
  v8f acc[4] = {{}, {}, {}, {}};
  for (int kb = 0; kb < DI; kb += 32) {
    v16h a;
#pragma unroll
    for (int i = 0; i < 16; ++i) a[i] = yg[(size_t)mrow * DI + kb + frag_k(lane, i)];
#pragma unroll
    for (int nt = 0; nt < 4; ++nt) {
      int ncol = (tng * 4 + nt) * 16 + (lane & 15);
      v16h bfr;
#pragma unroll
      for (int i = 0; i < 16; ++i) bfr[i] = wf[(size_t)ncol * DI + kb + frag_k(lane, i)];
      acc[nt] = __builtin_amdgcn_wmma_f32_16x16x32_f16(false, a, false, bfr, (short)0, acc[nt], false, false);
    }
  }
#pragma unroll
  for (int nt = 0; nt < 4; ++nt)
#pragma unroll
    for (int r = 0; r < 8; ++r) {
      int row = tm * 16 + r + ((lane & 16) ? 8 : 0);
      int col = (tng * 4 + nt) * 16 + (lane & 15);
      int b = row >> 12, hw = row & 4095;
      float res = input[((size_t)b * CC + col) * LL + hw] * skip[col];
      xafter[(size_t)row * CC + col] = acc[nt][r] + res;
    }
}

// ------------------------------------------------ LN2 over C (per pixel) ---
__global__ void k_ln2(const float* __restrict__ x, const float* __restrict__ g,
                      const float* __restrict__ bt, float* __restrict__ xln) {
  int m = blockIdx.x * blockDim.x + threadIdx.x;
  if (m >= MM) return;
  const float* row = x + (size_t)m * CC;
  float mu = 0.f;
  for (int c = 0; c < CC; ++c) mu += row[c];
  mu *= (1.f / CC);
  float var = 0.f;
  for (int c = 0; c < CC; ++c) { float t = row[c] - mu; var += t * t; }
  var *= (1.f / CC);
  float inv = rsqrtf(var + 1e-5f);
  for (int c = 0; c < CC; ++c)
    xln[(size_t)m * CC + c] = (row[c] - mu) * inv * g[c] + bt[c];
}

// ---------------------------------------- CAB conv1: 128->42 3x3 + GELU ---
__global__ void k_cab1(const float* __restrict__ xln, const float* __restrict__ w,
                       const float* __restrict__ bias, float* __restrict__ t1) {
  int idx = blockIdx.x * blockDim.x + threadIdx.x;       // (b,co,hw)
  if (idx >= BB * CMID * LL) return;
  int hw = idx & 4095;
  int co = (idx >> 12) % CMID;
  int b  = idx / (CMID * LL);
  int h = hw >> 6, wq = hw & 63;
  float s = bias[co];
  for (int i = 0; i < 3; ++i) {
    int hh = h + i - 1;
    if ((unsigned)hh >= 64u) continue;
    for (int j = 0; j < 3; ++j) {
      int ww = wq + j - 1;
      if ((unsigned)ww >= 64u) continue;
      const float* px = xln + ((size_t)b * LL + hh * 64 + ww) * CC;
      const float* wr = w + ((size_t)co * CC) * 9 + i * 3 + j;
      float acc = 0.f;
      for (int ci = 0; ci < CC; ++ci) acc += px[ci] * wr[(size_t)ci * 9];
      s += acc;
    }
  }
  t1[((size_t)b * CMID + co) * LL + hw] = geluf(s);
}

// ---------------------------------------- CAB conv2: 42->128 3x3 + bias ---
__global__ void k_cab2(const float* __restrict__ t1, const float* __restrict__ w,
                       const float* __restrict__ bias, float* __restrict__ t2) {
  int idx = blockIdx.x * blockDim.x + threadIdx.x;       // (b,co,hw)
  if (idx >= BB * CC * LL) return;
  int hw = idx & 4095, co = (idx >> 12) & 127, b = idx >> 19;
  int h = hw >> 6, wq = hw & 63;
  float s = bias[co];
  for (int ci = 0; ci < CMID; ++ci) {
    const float* src = t1 + ((size_t)b * CMID + ci) * LL;
    const float* wr  = w + ((size_t)co * CMID + ci) * 9;
#pragma unroll
    for (int i = 0; i < 3; ++i) {
      int hh = h + i - 1;
      if ((unsigned)hh >= 64u) continue;
#pragma unroll
      for (int j = 0; j < 3; ++j) {
        int ww = wq + j - 1;
        if ((unsigned)ww >= 64u) continue;
        s += src[hh * 64 + ww] * wr[i * 3 + j];
      }
    }
  }
  t2[idx] = s;
}

// ---------------------------------- channel-attention: global mean pool ---
__global__ void k_camean(const float* __restrict__ t2, float* __restrict__ smean) {
  __shared__ float red[256];
  int bc = blockIdx.x;                                   // b*128+c
  const float* src = t2 + (size_t)bc * LL;
  float s = 0.f;
  for (int i = threadIdx.x; i < LL; i += 256) s += src[i];
  red[threadIdx.x] = s;
  __syncthreads();
  for (int st = 128; st; st >>= 1) {
    if ((int)threadIdx.x < st) red[threadIdx.x] += red[threadIdx.x + st];
    __syncthreads();
  }
  if (threadIdx.x == 0) smean[bc] = red[0] * (1.f / LL);
}

__global__ void k_ca1(const float* __restrict__ smean, const float* __restrict__ w,
                      const float* __restrict__ bias, float* __restrict__ s1) {
  int t = threadIdx.x;
  if (t >= BB * CSQ) return;
  int b = t >> 2, o = t & 3;
  float s = bias[o];
  for (int c = 0; c < CC; ++c) s += w[(size_t)o * CC + c] * smean[b * CC + c];
  s1[t] = fmaxf(s, 0.f);
}

__global__ void k_ca2(const float* __restrict__ s1, const float* __restrict__ w,
                      const float* __restrict__ bias, float* __restrict__ s2) {
  int t = threadIdx.x;
  if (t >= BB * CC) return;
  int b = t >> 7, c = t & 127;
  float s = bias[c];
#pragma unroll
  for (int o = 0; o < CSQ; ++o) s += w[(size_t)c * CSQ + o] * s1[b * CSQ + o];
  s2[t] = 1.f / (1.f + __expf(-s));
}

// --------------- final: out(B,C,H,W) = x*skip2 + t2*s2  (with transpose) ---
__global__ void k_final(const float* __restrict__ xafter, const float* __restrict__ skip2,
                        const float* __restrict__ t2, const float* __restrict__ s2,
                        float* __restrict__ out) {
  int idx = blockIdx.x * blockDim.x + threadIdx.x;       // (b,c,hw)
  if (idx >= BB * CC * LL) return;
  int hw = idx & 4095, c = (idx >> 12) & 127, b = idx >> 19;
  float v = xafter[((size_t)b * LL + hw) * CC + c] * skip2[c] + t2[idx] * s2[b * CC + c];
  out[idx] = v;
}

// ===========================================================================
extern "C" void kernel_launch(void* const* d_in, const int* in_sizes, int n_in,
                              void* d_out, int out_size, void* d_ws, size_t ws_size,
                              hipStream_t stream) {
  const float* input    = (const float*)d_in[0];
  const float* ln1_g    = (const float*)d_in[1];
  const float* ln1_b    = (const float*)d_in[2];
  const float* skip1    = (const float*)d_in[3];
  const float* skip2    = (const float*)d_in[4];
  const float* ln2_g    = (const float*)d_in[5];
  const float* ln2_b    = (const float*)d_in[6];
  const float* in_proj  = (const float*)d_in[7];   // (512,128)
  const float* conv_w   = (const float*)d_in[8];   // (256,1,3,3)
  const float* conv_b   = (const float*)d_in[9];
  const float* x_proj_w = (const float*)d_in[10];  // (4,40,256)
  const float* dt_w     = (const float*)d_in[11];  // (4,256,8)
  const float* dt_b     = (const float*)d_in[12];  // (4,256)
  const float* A_logs   = (const float*)d_in[13];  // (1024,16)
  const float* Ds       = (const float*)d_in[14];  // (1024)
  const float* onorm_g  = (const float*)d_in[15];
  const float* onorm_b  = (const float*)d_in[16];
  const float* out_proj = (const float*)d_in[17];  // (128,256)
  const float* cab_w1   = (const float*)d_in[18];
  const float* cab_b1   = (const float*)d_in[19];
  const float* cab_w2   = (const float*)d_in[20];
  const float* cab_b2   = (const float*)d_in[21];
  const float* ca_w1    = (const float*)d_in[22];
  const float* ca_b1    = (const float*)d_in[23];
  const float* ca_w2    = (const float*)d_in[24];
  const float* ca_b2    = (const float*)d_in[25];
  float* out = (float*)d_out;

  // ---- workspace carve-up (~63 MB total) ----
  char* ws = (char*)d_ws;
  size_t off = 0;
  auto carve = [&](size_t bytes) { char* p = ws + off; off = (off + bytes + 255) & ~(size_t)255; return p; };
  _Float16* XN     = (_Float16*)carve((size_t)MM * CC * 2);      // ln1 output, f16
  _Float16* WIN16  = (_Float16*)carve((size_t)512 * CC * 2);
  _Float16* WOUT16 = (_Float16*)carve((size_t)CC * DI * 2);
  _Float16* WXP16  = (_Float16*)carve((size_t)KD * 40 * DI * 2);
  float*    XC     = (float*)carve((size_t)BB * DI * LL * 4);    // (B,D,H,W)
  float*    Z      = (float*)carve((size_t)MM * DI * 4);
  float*    XCA    = (float*)carve((size_t)BB * DI * LL * 4);    // silu(conv), f32 NCHW
  _Float16* XCAT   = (_Float16*)carve((size_t)MM * DI * 2);      // silu(conv), f16 NHWC
  float*    XDBL   = (float*)carve((size_t)BB * KD * 40 * LL * 4);
  float*    YCOMB  = (float*)carve((size_t)MM * DI * 4);
  _Float16* YG16   = (_Float16*)carve((size_t)MM * DI * 2);
  float*    XAFTER = (float*)carve((size_t)MM * CC * 4);
  float*    XLN2   = (float*)carve((size_t)MM * CC * 4);
  float*    T1     = (float*)carve((size_t)BB * CMID * LL * 4);
  float*    T2     = (float*)carve((size_t)BB * CC * LL * 4);
  float*    SMEAN  = (float*)carve((size_t)BB * CC * 4);
  float*    S1V    = (float*)carve((size_t)BB * CSQ * 4);
  float*    S2V    = (float*)carve((size_t)BB * CC * 4);
  (void)ws_size; (void)in_sizes; (void)n_in; (void)out_size;

  const int T = 256;
  // weight conversions to f16
  k_f32_to_f16<<<(512 * CC + T - 1) / T, T, 0, stream>>>(in_proj, WIN16, 512 * CC);
  k_f32_to_f16<<<(CC * DI + T - 1) / T, T, 0, stream>>>(out_proj, WOUT16, CC * DI);
  k_f32_to_f16<<<(KD * 40 * DI + T - 1) / T, T, 0, stream>>>(x_proj_w, WXP16, KD * 40 * DI);

  // LN1 -> f16
  k_ln1<<<MM / T, T, 0, stream>>>(input, ln1_g, ln1_b, XN);
  // in_proj GEMM (WMMA, 16x64 strips): 4096 waves
  k_inproj<<<512, T, 0, stream>>>(XN, WIN16, XC, Z);
  // depthwise conv + SiLU (+ f16 NHWC transpose for x_proj)
  k_dwconv<<<(BB * DI * LL) / T, T, 0, stream>>>(XC, conv_w, conv_b, XCA, XCAT);
  // x_proj GEMM (WMMA, 16x48 strips): 256 waves per (b,k)
  k_xproj<<<dim3(32, BB * KD), T, 0, stream>>>(XCAT, WXP16, XDBL);
  // zero combined-scan output, then scan
  k_zero<<<(MM * DI) / T, T, 0, stream>>>(YCOMB, MM * DI);
  k_scan<<<(BB * KD * DI * DSN) / T, T, 0, stream>>>(XCA, XDBL, dt_w, dt_b, A_logs, Ds, YCOMB);
  // out_norm + silu(z) gate -> f16
  k_gate<<<MM / T, T, 0, stream>>>(YCOMB, Z, onorm_g, onorm_b, YG16);
  // out_proj GEMM (WMMA, 16x64 strips) with fused skip residual
  k_outproj<<<128, T, 0, stream>>>(YG16, WOUT16, input, skip1, XAFTER);
  // CAB branch
  k_ln2<<<MM / T, T, 0, stream>>>(XAFTER, ln2_g, ln2_b, XLN2);
  k_cab1<<<(BB * CMID * LL + T - 1) / T, T, 0, stream>>>(XLN2, cab_w1, cab_b1, T1);
  k_cab2<<<(BB * CC * LL) / T, T, 0, stream>>>(T1, cab_w2, cab_b2, T2);
  k_camean<<<BB * CC, T, 0, stream>>>(T2, SMEAN);
  k_ca1<<<1, 32, 0, stream>>>(SMEAN, ca_w1, ca_b1, S1V);
  k_ca2<<<1, 256, 0, stream>>>(S1V, ca_w2, ca_b2, S2V);
  // final residual + channel attention + transpose to (B,C,H,W)
  k_final<<<(BB * CC * LL) / T, T, 0, stream>>>(XAFTER, skip2, T2, S2V, out);
}
